// VAE_LM_encoder_5463198401026
// MI455X (gfx1250) — matrix operand order repaired and checked by reading, not compiled, not run on hardware
//
#include <hip/hip_runtime.h>
#include <hip/hip_bf16.h>

// ---------------------------------------------------------------------------
// VAE-LM encoder for MI455X (gfx1250): bf16 WMMA everywhere.
//   Phase 1 (parallel): gather embed -> bf16, big GEMM X @ Wih0^T -> G0 (f32)
//   Phase 2 (sequential, 127 steps x 2 kernels): WMMA recurrent GEMMs with
//     4 gate accumulators per wave, LSTM cell applied in-register.
//     Inner loops are software-pipelined (prefetch k-tile t+1 fragments while
//     the WMMAs of tile t execute) with distinct fragment registers so the
//     compiler can issue partial loadcnt waits instead of full stalls.
//   Phase 3: VAE head (fc1+relu, mu/lv heads, reparameterize).
// Workspace requirement: ~122 MB (dominated by f32 G0 = 127*128*1536*4).
// ---------------------------------------------------------------------------

typedef __attribute__((ext_vector_type(16))) __bf16 v16bf;
typedef __attribute__((ext_vector_type(8)))  float  v8f;

union BF16Frag { unsigned u[8]; v16bf v; };

#define B_   128
#define T_   128
#define TS_  127
#define E_   512
#define H_   384
#define FH_  1536
#define ZH_  256
#define Z_   128

__device__ __forceinline__ unsigned short f2bf(float f) {
  unsigned u = __float_as_uint(f);
  unsigned r = u + 0x7fffu + ((u >> 16) & 1u);   // round-to-nearest-even
  return (unsigned short)(r >> 16);
}
__device__ __forceinline__ unsigned packbf(float a, float b) {
  return (unsigned)f2bf(a) | ((unsigned)f2bf(b) << 16);
}
__device__ __forceinline__ float sigm(float x) { return 1.0f / (1.0f + __expf(-x)); }

__device__ __forceinline__ v8f zero8() {
  v8f z;
#pragma unroll
  for (int i = 0; i < 8; ++i) z[i] = 0.0f;
  return z;
}

// A fragment: 16x32 bf16, row-major source [rows][ldk].
// Lane l: M = l&15.  VGPR v: K = hl*16+2v, hl = l>>4.
// Per lane this is two contiguous 16-byte runs -> compiler emits 2x b128.
__device__ __forceinline__ void load_a(const unsigned short* __restrict__ Abf, int ldk,
                                       int rowBase, int kBase, int lane, BF16Frag& fa) {
  int m = lane & 15, hl = lane >> 4;
  const unsigned short* p = Abf + (size_t)(rowBase + m) * ldk + kBase;
#pragma unroll
  for (int v = 0; v < 8; ++v) {
    int kk = hl * 16 + 2 * v;
    fa.u[v] = *(const unsigned*)(p + kk);
  }
}

// B fragment from pre-swizzled weights: per (jt,kt) 16x32 block, 256 dwords,
// lane l owns 8 contiguous dwords -> two b128 loads.
__device__ __forceinline__ void load_b(const unsigned* __restrict__ Wswz, int Ktiles,
                                       int jt, int kt, int lane, BF16Frag& fb) {
  const uint4* q = (const uint4*)(Wswz + ((size_t)(jt * Ktiles + kt) << 8) + (lane << 3));
  uint4 x0 = q[0], x1 = q[1];
  fb.u[0] = x0.x; fb.u[1] = x0.y; fb.u[2] = x0.z; fb.u[3] = x0.w;
  fb.u[4] = x1.x; fb.u[5] = x1.y; fb.u[6] = x1.z; fb.u[7] = x1.w;
}

__device__ __forceinline__ v8f wmma_bf16(const BF16Frag& fa, const BF16Frag& fb, v8f c) {
  return __builtin_amdgcn_wmma_f32_16x16x32_bf16(false, fa.v, false, fb.v, (short)0, c, false, false);
}

// Load one pipeline stage: A tile + 4 B tiles (jt = jtBase + g*jtStride).
__device__ __forceinline__ void load_stage(const unsigned short* __restrict__ Abf, int ldk, int rowBase,
                                           const unsigned* __restrict__ Wswz, int Ktiles,
                                           int jtBase, int jtStride, int kt, int lane,
                                           BF16Frag& fa, BF16Frag (&fb)[4]) {
  load_a(Abf, ldk, rowBase, kt * 32, lane, fa);
#pragma unroll
  for (int g = 0; g < 4; ++g) load_b(Wswz, Ktiles, jtBase + g * jtStride, kt, lane, fb[g]);
}

// ---------------- weight convert + swizzle (f32 row-major -> bf16 B-fragment layout)
// B layout (16x16x32 bf16): lanes 0-15 hold K=0..15, lanes 16-31 hold K=16..31,
// dword v of a lane packs K = hl*16 + 2v, +1;  N = lane&15 (= row j of W since B=W^T).
__global__ void k_convert_b(const float* __restrict__ W, unsigned* __restrict__ dst, int J, int K) {
  int d = blockIdx.x * blockDim.x + threadIdx.x;
  int total = (J * K) >> 1;
  if (d >= total) return;
  int Ktiles = K >> 5;
  int blk = d >> 8, within = d & 255;
  int lane = within >> 3, v = within & 7;
  int jt = blk / Ktiles, kt = blk % Ktiles;
  int j = jt * 16 + (lane & 15);
  int k = kt * 32 + (lane >> 4) * 16 + 2 * v;
  dst[d] = packbf(W[(size_t)j * K + k], W[(size_t)j * K + k + 1]);
}

// ---------------- embedding gather -> bf16 rows [TS_*B_][E_], row r = t*B_+b
__global__ void k_gather_embed(const long long* __restrict__ seq, const float* __restrict__ embed,
                               unsigned short* __restrict__ xbf) {
  int idx = blockIdx.x * blockDim.x + threadIdx.x;
  if (idx >= TS_ * B_ * E_) return;
  int r = idx / E_, k = idx - r * E_;
  int t = r >> 7, b = r & 127;
  long long tok = seq[(size_t)b * T_ + t + 1];
  xbf[idx] = f2bf(embed[(size_t)tok * E_ + k]);
}

// ---------------- sticky "active" flags (torch breaks at first all-zero step)
__global__ void k_active(const long long* __restrict__ seq, int* __restrict__ active) {
  __shared__ int nz[TS_];
  int t = threadIdx.x;
  if (t < TS_) {
    long long s = 0;
    for (int b = 0; b < B_; ++b) s += seq[(size_t)b * T_ + t + 1];
    nz[t] = (s != 0);
  }
  __syncthreads();
  if (t == 0) {
    int act = 1;
    for (int i = 0; i < TS_; ++i) { act = act && nz[i]; active[i] = act; }
  }
}

__global__ void k_zero(unsigned* __restrict__ p, int n) {
  int i = blockIdx.x * blockDim.x + threadIdx.x;
  if (i < n) p[i] = 0;
}

// ---------------- Phase 1 big GEMM: G0 = X @ Wih0^T ; [16256 x 1536], K=512
__global__ void k_gemm_x_wih0(const unsigned short* __restrict__ xbf,
                              const unsigned* __restrict__ Wswz, float* __restrict__ G0) {
  int wid = blockIdx.x * (blockDim.x >> 5) + (threadIdx.x >> 5);
  int lane = threadIdx.x & 31;
  const int RT = (TS_ * B_) / 16, CG = FH_ / 64;
  if (wid >= RT * CG) return;
  int rt = wid / CG, cg = wid % CG;
  v8f acc[4];
#pragma unroll
  for (int q = 0; q < 4; ++q) acc[q] = zero8();
  const int Ktiles = E_ / 32;                   // 16
  BF16Frag fa[2], fb[2][4];
  load_stage(xbf, E_, rt * 16, Wswz, Ktiles, cg * 4, 1, 0, lane, fa[0], fb[0]);
  for (int kt = 0; kt < Ktiles; ++kt) {
    int cur = kt & 1;
    if (kt + 1 < Ktiles)
      load_stage(xbf, E_, rt * 16, Wswz, Ktiles, cg * 4, 1, kt + 1, lane, fa[cur ^ 1], fb[cur ^ 1]);
#pragma unroll
    for (int q = 0; q < 4; ++q) acc[q] = wmma_bf16(fa[cur], fb[cur][q], acc[q]);
  }
  int hl = lane >> 4, n = lane & 15;
#pragma unroll
  for (int q = 0; q < 4; ++q) {
    int col = cg * 64 + q * 16 + n;
#pragma unroll
    for (int i = 0; i < 8; ++i)
      G0[(size_t)(rt * 16 + i + 8 * hl) * FH_ + col] = acc[q][i];
  }
}

// ---------------- layer-0 recurrent step: gates = G0[t] + h0_prev @ Whh0^T, cell
__global__ void k_step_l0(const float* __restrict__ G0, const unsigned* __restrict__ Whh0swz,
                          const unsigned short* __restrict__ h0_prev,
                          unsigned short* __restrict__ h0_cur, float* __restrict__ c0,
                          const int* __restrict__ active, int t) {
  if (!active[t]) return;
  int wid = blockIdx.x * (blockDim.x >> 5) + (threadIdx.x >> 5);
  int lane = threadIdx.x & 31;
  const int CT = H_ / 16;                       // 24
  if (wid >= 8 * CT) return;
  int rt = wid / CT, ct = wid % CT;
  int hl = lane >> 4, n = lane & 15;
  const int Ktiles = H_ / 32;                   // 12
  // Prefetch stage 0 before the (cached) G0 accumulator init.
  BF16Frag fa[2], fb[2][4];
  load_stage(h0_prev, H_, rt * 16, Whh0swz, Ktiles, ct, CT, 0, lane, fa[0], fb[0]);
  v8f acc[4];
#pragma unroll
  for (int g = 0; g < 4; ++g) {
#pragma unroll
    for (int i = 0; i < 8; ++i)
      acc[g][i] = G0[(size_t)(t * B_ + rt * 16 + i + 8 * hl) * FH_ + g * H_ + ct * 16 + n];
  }
  for (int kt = 0; kt < Ktiles; ++kt) {
    int cur = kt & 1;
    if (kt + 1 < Ktiles)
      load_stage(h0_prev, H_, rt * 16, Whh0swz, Ktiles, ct, CT, kt + 1, lane, fa[cur ^ 1], fb[cur ^ 1]);
#pragma unroll
    for (int g = 0; g < 4; ++g) acc[g] = wmma_bf16(fa[cur], fb[cur][g], acc[g]);
  }
#pragma unroll
  for (int i = 0; i < 8; ++i) {
    int b = rt * 16 + i + 8 * hl, j = ct * 16 + n;
    float cO = c0[b * H_ + j];
    float cN = sigm(acc[1][i]) * cO + sigm(acc[0][i]) * tanhf(acc[2][i]);
    float hN = sigm(acc[3][i]) * tanhf(cN);
    c0[b * H_ + j] = cN;
    h0_cur[b * H_ + j] = f2bf(hN);
  }
}

// ---------------- layer-1 recurrent step: gates = h0_cur@Wih1^T + h1_prev@Whh1^T, cell
// Both K-loops fused into one pipelined 2*Ktiles loop.
__global__ void k_step_l1(const unsigned* __restrict__ Wih1swz, const unsigned* __restrict__ Whh1swz,
                          const unsigned short* __restrict__ h0_cur,
                          const unsigned short* __restrict__ h1_prev,
                          unsigned short* __restrict__ h1_cur, unsigned short* __restrict__ h1fin,
                          float* __restrict__ c1, float* __restrict__ h1f,
                          const int* __restrict__ active, int t) {
  if (!active[t]) return;
  int wid = blockIdx.x * (blockDim.x >> 5) + (threadIdx.x >> 5);
  int lane = threadIdx.x & 31;
  const int CT = H_ / 16;
  if (wid >= 8 * CT) return;
  int rt = wid / CT, ct = wid % CT;
  int hl = lane >> 4, n = lane & 15;
  const int Ktiles = H_ / 32;
  v8f acc[4];
#pragma unroll
  for (int g = 0; g < 4; ++g) acc[g] = zero8();
  BF16Frag fa[2], fb[2][4];
  load_stage(h0_cur, H_, rt * 16, Wih1swz, Ktiles, ct, CT, 0, lane, fa[0], fb[0]);
  for (int s = 0; s < 2 * Ktiles; ++s) {
    int cur = s & 1;
    int sn = s + 1;
    if (sn < 2 * Ktiles) {
      const unsigned short* A = (sn < Ktiles) ? h0_cur : h1_prev;
      const unsigned* W = (sn < Ktiles) ? Wih1swz : Whh1swz;
      int kt = (sn < Ktiles) ? sn : sn - Ktiles;
      load_stage(A, H_, rt * 16, W, Ktiles, ct, CT, kt, lane, fa[cur ^ 1], fb[cur ^ 1]);
    }
#pragma unroll
    for (int g = 0; g < 4; ++g) acc[g] = wmma_bf16(fa[cur], fb[cur][g], acc[g]);
  }
#pragma unroll
  for (int i = 0; i < 8; ++i) {
    int b = rt * 16 + i + 8 * hl, j = ct * 16 + n;
    float cO = c1[b * H_ + j];
    float cN = sigm(acc[1][i]) * cO + sigm(acc[0][i]) * tanhf(acc[2][i]);
    float hN = sigm(acc[3][i]) * tanhf(cN);
    c1[b * H_ + j] = cN;
    unsigned short hb = f2bf(hN);
    h1_cur[b * H_ + j] = hb;
    h1fin[b * H_ + j] = hb;
    h1f[b * H_ + j] = hN;
  }
}

// ---------------- head: hv = relu(rnn @ W1^T + b1)  [128 x 256], K=384
__global__ void k_head_fc1(const unsigned short* __restrict__ h1bf, const unsigned* __restrict__ W1swz,
                           const float* __restrict__ b1, unsigned short* __restrict__ hvbf) {
  int wid = blockIdx.x * (blockDim.x >> 5) + (threadIdx.x >> 5);
  int lane = threadIdx.x & 31;
  const int CT = ZH_ / 16;                      // 16
  if (wid >= 8 * CT) return;
  int rt = wid / CT, ct = wid % CT;
  int hl = lane >> 4, n = lane & 15;
  float bias = b1[ct * 16 + n];
  v8f acc;
#pragma unroll
  for (int i = 0; i < 8; ++i) acc[i] = bias;
  const int Ktiles = H_ / 32;
  BF16Frag fa[2], fb[2];
  load_a(h1bf, H_, rt * 16, 0, lane, fa[0]);
  load_b(W1swz, Ktiles, ct, 0, lane, fb[0]);
  for (int kt = 0; kt < Ktiles; ++kt) {
    int cur = kt & 1;
    if (kt + 1 < Ktiles) {
      load_a(h1bf, H_, rt * 16, (kt + 1) * 32, lane, fa[cur ^ 1]);
      load_b(W1swz, Ktiles, ct, kt + 1, lane, fb[cur ^ 1]);
    }
    acc = wmma_bf16(fa[cur], fb[cur], acc);
  }
#pragma unroll
  for (int i = 0; i < 8; ++i) {
    int b = rt * 16 + i + 8 * hl, j = ct * 16 + n;
    hvbf[b * ZH_ + j] = f2bf(fmaxf(acc[i], 0.0f));
  }
}

// ---------------- head: z_mu, z_lv, z = mu + eps*exp(0.5*lv)  [128 x 128], K=256
__global__ void k_head_z(const unsigned short* __restrict__ hvbf,
                         const unsigned* __restrict__ Wmuswz, const unsigned* __restrict__ Wlvswz,
                         const float* __restrict__ bmu, const float* __restrict__ blv,
                         const float* __restrict__ eps,
                         float* __restrict__ out_mu, float* __restrict__ out_lv,
                         float* __restrict__ out_cat) {
  int wid = blockIdx.x * (blockDim.x >> 5) + (threadIdx.x >> 5);
  int lane = threadIdx.x & 31;
  const int CT = Z_ / 16;                       // 8
  if (wid >= 8 * CT) return;
  int rt = wid / CT, ct = wid % CT;
  int hl = lane >> 4, n = lane & 15;
  float bm = bmu[ct * 16 + n], bl = blv[ct * 16 + n];
  v8f am, al;
#pragma unroll
  for (int i = 0; i < 8; ++i) { am[i] = bm; al[i] = bl; }
  const int Ktiles = ZH_ / 32;                  // 8
  BF16Frag fa[2], fbm[2], fbl[2];
  load_a(hvbf, ZH_, rt * 16, 0, lane, fa[0]);
  load_b(Wmuswz, Ktiles, ct, 0, lane, fbm[0]);
  load_b(Wlvswz, Ktiles, ct, 0, lane, fbl[0]);
  for (int kt = 0; kt < Ktiles; ++kt) {
    int cur = kt & 1;
    if (kt + 1 < Ktiles) {
      load_a(hvbf, ZH_, rt * 16, (kt + 1) * 32, lane, fa[cur ^ 1]);
      load_b(Wmuswz, Ktiles, ct, kt + 1, lane, fbm[cur ^ 1]);
      load_b(Wlvswz, Ktiles, ct, kt + 1, lane, fbl[cur ^ 1]);
    }
    am = wmma_bf16(fa[cur], fbm[cur], am);
    al = wmma_bf16(fa[cur], fbl[cur], al);
  }
#pragma unroll
  for (int i = 0; i < 8; ++i) {
    int b = rt * 16 + i + 8 * hl, j = ct * 16 + n;
    float mu = am[i], lv = al[i];
    float z = mu + eps[b * Z_ + j] * __expf(0.5f * lv);
    out_mu[b * Z_ + j] = mu;
    out_lv[b * Z_ + j] = lv;
    out_cat[b * (H_ + Z_) + H_ + j] = z;
  }
}

__global__ void k_copy_rnn(const float* __restrict__ h1f, float* __restrict__ out_cat) {
  int i = blockIdx.x * blockDim.x + threadIdx.x;
  if (i < B_ * H_) {
    int b = i / H_, j = i - b * H_;
    out_cat[b * (H_ + Z_) + j] = h1f[i];
  }
}

// ---------------------------------------------------------------------------
extern "C" void kernel_launch(void* const* d_in, const int* in_sizes, int n_in,
                              void* d_out, int out_size, void* d_ws, size_t ws_size,
                              hipStream_t stream) {
  const long long* seq = (const long long*)d_in[0];   // int64 [B,T]
  const float* eps   = (const float*)d_in[1];
  const float* embed = (const float*)d_in[2];
  const float* Wih0  = (const float*)d_in[3];
  const float* Whh0  = (const float*)d_in[4];
  const float* Wih1  = (const float*)d_in[5];
  const float* Whh1  = (const float*)d_in[6];
  const float* W1    = (const float*)d_in[7];
  const float* b1    = (const float*)d_in[8];
  const float* Wmu   = (const float*)d_in[9];
  const float* bmu   = (const float*)d_in[10];
  const float* Wlv   = (const float*)d_in[11];
  const float* blv   = (const float*)d_in[12];

  float* out     = (float*)d_out;
  float* out_mu  = out;
  float* out_lv  = out + B_ * Z_;
  float* out_cat = out + 2 * B_ * Z_;            // [B][H+Z]

  char* ws = (char*)d_ws;
  size_t off = 0;
  auto alloc = [&](size_t bytes) -> void* {
    void* p = ws + off;
    off += (bytes + 255) & ~(size_t)255;
    return p;
  };

  unsigned short* xbf     = (unsigned short*)alloc((size_t)TS_ * B_ * E_ * 2);
  float*          G0      = (float*)alloc((size_t)TS_ * B_ * FH_ * 4);
  unsigned* Wih0swz = (unsigned*)alloc((size_t)FH_ * E_ * 2);
  unsigned* Whh0swz = (unsigned*)alloc((size_t)FH_ * H_ * 2);
  unsigned* Wih1swz = (unsigned*)alloc((size_t)FH_ * H_ * 2);
  unsigned* Whh1swz = (unsigned*)alloc((size_t)FH_ * H_ * 2);
  unsigned* W1swz   = (unsigned*)alloc((size_t)ZH_ * H_ * 2);
  unsigned* Wmuswz  = (unsigned*)alloc((size_t)Z_ * ZH_ * 2);
  unsigned* Wlvswz  = (unsigned*)alloc((size_t)Z_ * ZH_ * 2);
  float* c0 = (float*)alloc((size_t)B_ * H_ * 4);
  float* c1 = (float*)alloc((size_t)B_ * H_ * 4);
  unsigned short* h0bf  = (unsigned short*)alloc((size_t)2 * B_ * H_ * 2);   // ping-pong
  unsigned short* h1bf  = (unsigned short*)alloc((size_t)2 * B_ * H_ * 2);   // ping-pong
  unsigned short* h1fin = (unsigned short*)alloc((size_t)B_ * H_ * 2);
  float*          h1f   = (float*)alloc((size_t)B_ * H_ * 4);
  unsigned short* hvbf  = (unsigned short*)alloc((size_t)B_ * ZH_ * 2);
  int* active = (int*)alloc(TS_ * sizeof(int));

  auto cdiv = [](int a, int b) { return (a + b - 1) / b; };

  // Weight conversion + swizzle into WMMA B-fragment layout.
  k_convert_b<<<cdiv(FH_ * E_ / 2, 256), 256, 0, stream>>>(Wih0, Wih0swz, FH_, E_);
  k_convert_b<<<cdiv(FH_ * H_ / 2, 256), 256, 0, stream>>>(Whh0, Whh0swz, FH_, H_);
  k_convert_b<<<cdiv(FH_ * H_ / 2, 256), 256, 0, stream>>>(Wih1, Wih1swz, FH_, H_);
  k_convert_b<<<cdiv(FH_ * H_ / 2, 256), 256, 0, stream>>>(Whh1, Whh1swz, FH_, H_);
  k_convert_b<<<cdiv(ZH_ * H_ / 2, 256), 256, 0, stream>>>(W1, W1swz, ZH_, H_);
  k_convert_b<<<cdiv(Z_ * ZH_ / 2, 256), 256, 0, stream>>>(Wmu, Wmuswz, Z_, ZH_);
  k_convert_b<<<cdiv(Z_ * ZH_ / 2, 256), 256, 0, stream>>>(Wlv, Wlvswz, Z_, ZH_);

  // State init (workspace is poisoned, not zeroed).
  k_zero<<<cdiv(B_ * H_, 256), 256, 0, stream>>>((unsigned*)c0, B_ * H_);
  k_zero<<<cdiv(B_ * H_, 256), 256, 0, stream>>>((unsigned*)c1, B_ * H_);
  k_zero<<<cdiv(B_ * H_, 256), 256, 0, stream>>>((unsigned*)h0bf, B_ * H_);      // both parities
  k_zero<<<cdiv(B_ * H_, 256), 256, 0, stream>>>((unsigned*)h1bf, B_ * H_);
  k_zero<<<cdiv(B_ * H_ / 2, 256), 256, 0, stream>>>((unsigned*)h1fin, B_ * H_ / 2);
  k_zero<<<cdiv(B_ * H_, 256), 256, 0, stream>>>((unsigned*)h1f, B_ * H_);

  // Sticky active flags + embedding gather.
  k_active<<<1, 128, 0, stream>>>(seq, active);
  k_gather_embed<<<cdiv(TS_ * B_ * E_, 256), 256, 0, stream>>>(seq, embed, xbf);

  // Phase 1: all-timesteps input GEMM (WMMA), 1016x24 wave tiles.
  {
    const int waves = ((TS_ * B_) / 16) * (FH_ / 64);
    k_gemm_x_wih0<<<cdiv(waves, 8), 256, 0, stream>>>(xbf, Wih0swz, G0);
  }

  // Phase 2: sequential recurrence; ping-pong h buffers by parity of t.
  const size_t hstride = (size_t)B_ * H_;
  for (int t = 0; t < TS_; ++t) {
    unsigned short* h0p = h0bf + (t & 1) * hstride;
    unsigned short* h0c = h0bf + ((t + 1) & 1) * hstride;
    unsigned short* h1p = h1bf + (t & 1) * hstride;
    unsigned short* h1c = h1bf + ((t + 1) & 1) * hstride;
    k_step_l0<<<24, 256, 0, stream>>>(G0, Whh0swz, h0p, h0c, c0, active, t);
    k_step_l1<<<24, 256, 0, stream>>>(Wih1swz, Whh1swz, h0c, h1p, h1c, h1fin, c1, h1f, active, t);
  }

  // Phase 3: VAE head.
  k_head_fc1<<<16, 256, 0, stream>>>(h1fin, W1swz, b1, hvbf);
  k_head_z<<<8, 256, 0, stream>>>(hvbf, Wmuswz, Wlvswz, bmu, blv, eps, out_mu, out_lv, out_cat);
  k_copy_rnn<<<cdiv(B_ * H_, 256), 256, 0, stream>>>(h1f, out_cat);

  (void)in_sizes; (void)n_in; (void)out_size; (void)ws_size;
}
